// TabularTransformerWithAttention_23364622090871
// MI455X (gfx1250) — compile-verified, matrix operand order checked by reading, not run on hardware
//
#include <hip/hip_runtime.h>
#include <hip/hip_bf16.h>
#include <cstdint>

// ---------------------------------------------------------------------------
// TabularTransformer on gfx1250 (CDNA5, wave32, WMMA).
// All matmuls use v_wmma_f32_16x16x32_f16 (f16 A/B, f32 accum).
// Attention is fused per (batch, 16-query-row tile) using ~260KB of LDS.
// GEMM A-panel staging uses the Tensor Data Mover (tensor_load_to_lds)
// when the builtin is available, with vectorized LDS staging as fallback.
// ---------------------------------------------------------------------------

typedef _Float16 half_t;
typedef __attribute__((ext_vector_type(16))) _Float16 v16h;
typedef __attribute__((ext_vector_type(8)))  float    v8f;
typedef __attribute__((ext_vector_type(4)))  unsigned int u32x4;
typedef __attribute__((ext_vector_type(8)))  int          i32x8;
typedef __attribute__((ext_vector_type(4)))  int          i32x4;

#define BATCH   8
#define SEQ     2048
#define DMODEL  128
#define NHEAD   4
#define DHEAD   32
#define FFDIM   512
#define MROWS   (BATCH * SEQ)       // 16384
#define QKVDIM  (3 * DMODEL)        // 384

__device__ __forceinline__ v8f wmma16x16x32(v16h a, v16h b, v8f c) {
  // (neg_a, A, neg_b, B, c_mod, C, reuse_a, reuse_b)
  return __builtin_amdgcn_wmma_f32_16x16x32_f16(false, a, false, b, (short)0, c,
                                                false, false);
}

// ---------------------------------------------------------------------------
// f32 -> f16 cast (weights)
// ---------------------------------------------------------------------------
__global__ void cast_f32_to_f16_kernel(const float* __restrict__ in,
                                       half_t* __restrict__ out, int n) {
  int i = blockIdx.x * 256 + threadIdx.x;
  if (i < n) out[i] = (half_t)in[i];
}

// ---------------------------------------------------------------------------
// Embedding: x[m,d] = table lookup + values[m]*val_w[d] + val_b[d]
// ---------------------------------------------------------------------------
__global__ void embed_kernel(const int* __restrict__ row_ids,
                             const int* __restrict__ col_ids,
                             const float* __restrict__ values,
                             const float* __restrict__ row_table,
                             const float* __restrict__ col_table,
                             const float* __restrict__ val_w,
                             const float* __restrict__ val_b,
                             float* __restrict__ xf, half_t* __restrict__ xh) {
  int i = blockIdx.x * 256 + threadIdx.x;
  if (i >= MROWS * DMODEL) return;
  int m = i >> 7, d = i & 127;
  float pos = (d < 64) ? row_table[row_ids[m] * 64 + d]
                       : col_table[col_ids[m] * 64 + (d - 64)];
  float v = pos + values[m] * val_w[d] + val_b[d];
  xf[i] = v;
  xh[i] = (half_t)v;
}

// ---------------------------------------------------------------------------
// V transpose: vT[b,h,dh,s] = qkv[b*S+s, 256 + h*32 + dh]
// Makes attention phase-C B-fragments contiguous 16B loads.
// ---------------------------------------------------------------------------
__global__ void vtrans_kernel(const half_t* __restrict__ qkv,
                              half_t* __restrict__ vT) {
  int i = blockIdx.x * 256 + threadIdx.x;   // B*H*DHEAD*SEQ, s fastest
  if (i >= BATCH * NHEAD * DHEAD * SEQ) return;
  int s  = i % SEQ;
  int t  = i / SEQ;
  int dh = t % DHEAD;
  int bh = t / DHEAD;
  int h  = bh % NHEAD, b = bh / NHEAD;
  vT[i] = qkv[((size_t)b * SEQ + s) * QKVDIM + 2 * DMODEL + h * DHEAD + dh];
}

// ---------------------------------------------------------------------------
// Generic WMMA GEMM:  C[M,N] = A[M,K] @ W[N,K]^T + bias  (optional ReLU)
// Block = 256 threads (8 waves). Block tile: 16 rows x 128 cols.
// A panel (16 x K, K<=512) staged in LDS via TDM (or vector loads).
// ---------------------------------------------------------------------------
__global__ __launch_bounds__(256) void gemm_wmma_kernel(
    const half_t* __restrict__ A, const half_t* __restrict__ W,
    const float* __restrict__ bias, float* __restrict__ Cf,
    half_t* __restrict__ Ch, int M, int N, int K, int relu) {
  __shared__ half_t ldsA[16 * 512];  // 16KB, K <= 512

  int m0 = blockIdx.x * 16;
  int n0 = blockIdx.y * 128;
  int tid = threadIdx.x, wave = tid >> 5, lane = tid & 31;

#if defined(__gfx1250__) && __has_builtin(__builtin_amdgcn_tensor_load_to_lds)
  // ---- Tensor Data Mover: DMA 2D tile [16 rows x K halves] into LDS ----
  if (tid < 32) {  // single wave issues the TDM op (EXEC-independent)
    unsigned long long ga =
        (unsigned long long)(uintptr_t)(const char*)&A[(size_t)m0 * K];
    unsigned lds_off = (unsigned)(uintptr_t)&ldsA[0];
    unsigned Kd = (unsigned)K, Md = (unsigned)M;
    u32x4 g0;
    g0[0] = 1u;                                    // count=1 valid descriptor
    g0[1] = lds_off;                               // lds_addr
    g0[2] = (unsigned)(ga & 0xffffffffu);          // global_addr[31:0]
    g0[3] = (unsigned)((ga >> 32) & 0x01ffffffu) | 0x80000000u;  // [56:32]|type=2
    i32x8 g1;
    g1[0] = (int)(1u << 16);                       // data_size=1 (2 bytes)
    g1[1] = (int)((Kd & 0xffffu) << 16);           // tensor_dim0[15:0]
    g1[2] = (int)(((Kd >> 16) & 0xffffu) | ((Md & 0xffffu) << 16));
    g1[3] = (int)((((Md >> 16) & 0xffffu)) | ((Kd & 0xffffu) << 16)); // tile_dim0=K
    g1[4] = 16;                                    // tile_dim1=16
    g1[5] = (int)Kd;                               // tensor_dim0_stride=K
    g1[6] = 0;
    g1[7] = 0;
    i32x4 z4 = {0, 0, 0, 0};
#if __clang_major__ >= 23
    i32x8 z8 = {0, 0, 0, 0, 0, 0, 0, 0};
    __builtin_amdgcn_tensor_load_to_lds(g0, g1, z4, z4, z8, 0);
#else
    __builtin_amdgcn_tensor_load_to_lds(g0, g1, z4, z4, 0);
#endif
    __builtin_amdgcn_s_wait_tensorcnt(0);
  }
#else
  // ---- fallback: vectorized cooperative staging ----
  int kvec = K >> 3;  // 16B chunks per row
  for (int i = tid; i < 16 * kvec; i += 256) {
    int r = i / kvec, c = (i % kvec) * 8;
    *(uint4*)&ldsA[r * K + c] = *(const uint4*)&A[(size_t)(m0 + r) * K + c];
  }
#endif
  __syncthreads();

  const int row  = lane & 15;
  const int koff = (lane >> 4) * 8;
  const int ncol = n0 + wave * 16 + (lane & 15);

  v8f acc = {};
  for (int k0 = 0; k0 < K; k0 += 32) {
    v16h a, b;
    const half_t* ap = &ldsA[row * K + k0 + koff];
    const half_t* wp = &W[(size_t)ncol * K + k0 + koff];
    if (k0 + 32 < K) __builtin_prefetch(wp + 32, 0, 0);
#pragma unroll
    for (int j = 0; j < 8; ++j) {
      a[j]     = ap[j];
      a[8 + j] = ap[16 + j];
      b[j]     = wp[j];
      b[8 + j] = wp[16 + j];
    }
    acc = wmma16x16x32(a, b, acc);
  }

  float bv = bias ? bias[ncol] : 0.0f;
  int mbase = m0 + ((lane >> 4) * 8);
#pragma unroll
  for (int r = 0; r < 8; ++r) {
    float v = acc[r] + bv;
    if (relu) v = v > 0.0f ? v : 0.0f;
    size_t idx = (size_t)(mbase + r) * N + ncol;
    if (Cf) Cf[idx] = v;
    if (Ch) Ch[idx] = (half_t)v;
  }
}

// ---------------------------------------------------------------------------
// Fused attention: one block per (batch b, 16-query-row tile).
// LDS: sc[16][SEQ] scores/probs, avg[16][SEQ] head-average accumulator,
//      reductions, ctx[16][32]. Total ~260KB dynamic LDS (<=320KB WGP).
// qkv layout: [B*S, 384] f16 ; per head h: q@h*32, k@128+h*32.
// vT layout: [B,H,DHEAD,SEQ] f16 (pre-transposed V).
// ---------------------------------------------------------------------------
__global__ __launch_bounds__(256) void attn_fused_kernel(
    const half_t* __restrict__ qkv, const half_t* __restrict__ vT,
    half_t* __restrict__ ctx16, float* __restrict__ attn_out) {
  extern __shared__ float lds[];
  float* sc   = lds;                 // 16*SEQ
  float* avg  = sc + 16 * SEQ;       // 16*SEQ
  float* redm = avg + 16 * SEQ;      // 16*16
  float* reds = redm + 256;          // 16*16
  float* ctx  = reds + 256;          // 16*32

  const int nqt = SEQ / 16;
  const int b  = blockIdx.x / nqt;
  const int qt = blockIdx.x % nqt;
  const size_t rowbase = (size_t)b * SEQ + qt * 16;

  const int tid = threadIdx.x, wave = tid >> 5, lane = tid & 31;
  const int row  = lane & 15;
  const int koff = (lane >> 4) * 8;
  const float scale = 0.17677669529663687f;  // 1/sqrt(32)

  for (int i = tid; i < 16 * SEQ; i += 256) avg[i] = 0.0f;
  __syncthreads();

  for (int h = 0; h < NHEAD; ++h) {
    // ---- Phase A: scores[16][SEQ] = (q @ k^T) * scale via WMMA ----
    v16h aq;
    {
      const half_t* qp = &qkv[(rowbase + row) * QKVDIM + h * DHEAD + koff];
#pragma unroll
      for (int j = 0; j < 8; ++j) { aq[j] = qp[j]; aq[8 + j] = qp[16 + j]; }
    }
    for (int t = 0; t < (SEQ / 16) / 8; ++t) {  // 16 key tiles per wave
      int n = wave * (SEQ / 8) + t * 16 + (lane & 15);
      v16h bk;
      const half_t* kp =
          &qkv[((size_t)b * SEQ + n) * QKVDIM + DMODEL + h * DHEAD + koff];
#pragma unroll
      for (int j = 0; j < 8; ++j) { bk[j] = kp[j]; bk[8 + j] = kp[16 + j]; }
      v8f c = {};
      c = wmma16x16x32(aq, bk, c);
      int mb = (lane >> 4) * 8;
#pragma unroll
      for (int r = 0; r < 8; ++r) sc[(mb + r) * SEQ + n] = c[r] * scale;
    }
    __syncthreads();

    // ---- Phase B: row softmax over SEQ, accumulate 0.25*p into avg ----
    {
      int r = tid >> 4, cch = tid & 15;
      int c0 = cch * (SEQ / 16);
      float mx = -1e30f;
      for (int c = 0; c < SEQ / 16; ++c) mx = fmaxf(mx, sc[r * SEQ + c0 + c]);
      redm[r * 16 + cch] = mx;
      __syncthreads();
      float rmax = redm[r * 16];
#pragma unroll
      for (int i = 1; i < 16; ++i) rmax = fmaxf(rmax, redm[r * 16 + i]);
      float sum = 0.0f;
      for (int c = 0; c < SEQ / 16; ++c) {
        float e = __expf(sc[r * SEQ + c0 + c] - rmax);
        sc[r * SEQ + c0 + c] = e;
        sum += e;
      }
      reds[r * 16 + cch] = sum;
      __syncthreads();
      float tot = 0.0f;
#pragma unroll
      for (int i = 0; i < 16; ++i) tot += reds[r * 16 + i];
      float inv = 1.0f / tot;
      for (int c = 0; c < SEQ / 16; ++c) {
        float p = sc[r * SEQ + c0 + c] * inv;
        sc[r * SEQ + c0 + c] = p;
        avg[r * SEQ + c0 + c] += 0.25f * p;
      }
    }
    __syncthreads();
    for (int i = tid; i < 16 * 32; i += 256) ctx[i] = 0.0f;
    __syncthreads();

    // ---- Phase C: ctx[16][32] = p[16][SEQ] @ v[SEQ][32] via WMMA ----
    {
      v8f c0 = {}, c1 = {};
      int kbase = wave * (SEQ / 8);
      const half_t* vbase = &vT[((size_t)(b * NHEAD + h) * DHEAD) * SEQ];
      int dv = lane & 15;
      for (int ks = 0; ks < SEQ / 8; ks += 32) {
        int k0 = kbase + ks;
        v16h ap, bv0, bv1;
        const float* pp = &sc[row * SEQ + k0 + koff];
        const half_t* vp0 = vbase + (size_t)dv * SEQ + k0 + koff;
        const half_t* vp1 = vbase + (size_t)(16 + dv) * SEQ + k0 + koff;
#pragma unroll
        for (int j = 0; j < 8; ++j) {
          ap[j]      = (half_t)pp[j];
          ap[8 + j]  = (half_t)pp[16 + j];
          bv0[j]     = vp0[j];
          bv0[8 + j] = vp0[16 + j];
          bv1[j]     = vp1[j];
          bv1[8 + j] = vp1[16 + j];
        }
        c0 = wmma16x16x32(ap, bv0, c0);
        c1 = wmma16x16x32(ap, bv1, c1);
      }
      int mb = (lane >> 4) * 8, n = lane & 15;
#pragma unroll
      for (int r = 0; r < 8; ++r) {
        atomicAdd(&ctx[(mb + r) * 32 + n], c0[r]);          // ds_add_f32
        atomicAdd(&ctx[(mb + r) * 32 + 16 + n], c1[r]);
      }
    }
    __syncthreads();

    // write this head's context slice to f16 ctx buffer
    for (int i = tid; i < 16 * 32; i += 256) {
      int m = i >> 5, d = i & 31;
      ctx16[(rowbase + m) * DMODEL + h * DHEAD + d] = (half_t)ctx[m * 32 + d];
    }
    __syncthreads();
  }

  // ---- write head-averaged attention strip to output ----
  for (int i = tid; i < 16 * SEQ; i += 256) {
    int r = i / SEQ, c = i % SEQ;
    attn_out[((size_t)b * SEQ + qt * 16 + r) * SEQ + c] = avg[i];
  }
}

// ---------------------------------------------------------------------------
// Residual add + LayerNorm over D=128; one wave32 per row; writes f32 + f16.
// ---------------------------------------------------------------------------
__global__ __launch_bounds__(256) void add_ln_kernel(
    const float* __restrict__ resid, const float* __restrict__ y,
    const float* __restrict__ gamma, const float* __restrict__ beta,
    float* __restrict__ xf, half_t* __restrict__ xh) {
  int wave = threadIdx.x >> 5, lane = threadIdx.x & 31;
  int m = blockIdx.x * 8 + wave;
  const float* rp = resid + (size_t)m * DMODEL;
  const float* yp = y + (size_t)m * DMODEL;
  float v[4];
  float s = 0.0f;
#pragma unroll
  for (int j = 0; j < 4; ++j) {
    v[j] = rp[lane * 4 + j] + yp[lane * 4 + j];
    s += v[j];
  }
#pragma unroll
  for (int off = 16; off > 0; off >>= 1) s += __shfl_down(s, off, 32);
  s = __shfl(s, 0, 32);
  float mean = s * (1.0f / DMODEL);
  float var = 0.0f;
#pragma unroll
  for (int j = 0; j < 4; ++j) { float d = v[j] - mean; var += d * d; }
#pragma unroll
  for (int off = 16; off > 0; off >>= 1) var += __shfl_down(var, off, 32);
  var = __shfl(var, 0, 32) * (1.0f / DMODEL);
  float inv = rsqrtf(var + 1e-5f);
#pragma unroll
  for (int j = 0; j < 4; ++j) {
    int d = lane * 4 + j;
    float o = (v[j] - mean) * inv * gamma[d] + beta[d];
    xf[(size_t)m * DMODEL + d] = o;
    xh[(size_t)m * DMODEL + d] = (half_t)o;
  }
}

// ---------------------------------------------------------------------------
// predictions[m] = dot(x[m,:], w[:]) + b ; one wave32 per row.
// ---------------------------------------------------------------------------
__global__ __launch_bounds__(256) void predict_kernel(
    const float* __restrict__ xf, const float* __restrict__ w,
    const float* __restrict__ bptr, float* __restrict__ out) {
  int wave = threadIdx.x >> 5, lane = threadIdx.x & 31;
  int m = blockIdx.x * 8 + wave;
  const float* xp = xf + (size_t)m * DMODEL;
  float s = 0.0f;
#pragma unroll
  for (int j = 0; j < 4; ++j) s += xp[lane * 4 + j] * w[lane * 4 + j];
#pragma unroll
  for (int off = 16; off > 0; off >>= 1) s += __shfl_down(s, off, 32);
  if (lane == 0) out[m] = s + bptr[0];
}

// ---------------------------------------------------------------------------
extern "C" void kernel_launch(void* const* d_in, const int* in_sizes, int n_in,
                              void* d_out, int out_size, void* d_ws,
                              size_t ws_size, hipStream_t stream) {
  (void)in_sizes; (void)n_in; (void)out_size; (void)ws_size;

  const int*   row_ids  = (const int*)d_in[0];
  const int*   col_ids  = (const int*)d_in[1];
  const float* values   = (const float*)d_in[2];
  const float* row_tab  = (const float*)d_in[3];
  const float* col_tab  = (const float*)d_in[4];
  const float* val_w    = (const float*)d_in[5];
  const float* val_b    = (const float*)d_in[6];
  const float* in_w     = (const float*)d_in[7];    // [2,384,128]
  const float* in_b     = (const float*)d_in[8];    // [2,384]
  const float* out_w    = (const float*)d_in[9];    // [2,128,128]
  const float* out_b    = (const float*)d_in[10];   // [2,128]
  const float* lin1_w   = (const float*)d_in[11];   // [2,512,128]
  const float* lin1_b   = (const float*)d_in[12];   // [2,512]
  const float* lin2_w   = (const float*)d_in[13];   // [2,128,512]
  const float* lin2_b   = (const float*)d_in[14];   // [2,128]
  const float* ln1_g    = (const float*)d_in[15];
  const float* ln1_b    = (const float*)d_in[16];
  const float* ln2_g    = (const float*)d_in[17];
  const float* ln2_b    = (const float*)d_in[18];
  const float* outp_w   = (const float*)d_in[19];
  const float* outp_b   = (const float*)d_in[20];

  // ---- workspace carve ----
  size_t off = 0;
  auto carve = [&](size_t bytes) {
    void* p = (char*)d_ws + off;
    off += (bytes + 255) & ~(size_t)255;
    return p;
  };
  float*  xf      = (float*)carve((size_t)MROWS * DMODEL * 4);
  half_t* xh      = (half_t*)carve((size_t)MROWS * DMODEL * 2);
  half_t* qkvh    = (half_t*)carve((size_t)MROWS * QKVDIM * 2);
  half_t* vT      = (half_t*)carve((size_t)BATCH * NHEAD * DHEAD * SEQ * 2);
  half_t* ffh     = (half_t*)carve((size_t)MROWS * FFDIM * 2);
  half_t* ctx16   = (half_t*)carve((size_t)MROWS * DMODEL * 2);
  float*  ybuf    = (float*)carve((size_t)MROWS * DMODEL * 4);
  half_t* in_w16  = (half_t*)carve((size_t)2 * QKVDIM * DMODEL * 2);
  half_t* out_w16 = (half_t*)carve((size_t)2 * DMODEL * DMODEL * 2);
  half_t* l1_w16  = (half_t*)carve((size_t)2 * FFDIM * DMODEL * 2);
  half_t* l2_w16  = (half_t*)carve((size_t)2 * DMODEL * FFDIM * 2);

  float* pred  = (float*)d_out;
  float* attn0 = pred + MROWS;                           // [8,2048,2048]
  float* attn1 = attn0 + (size_t)BATCH * SEQ * SEQ;

  // ---- weight casts ----
  auto cast = [&](const float* src, half_t* dst, int n) {
    cast_f32_to_f16_kernel<<<(n + 255) / 256, 256, 0, stream>>>(src, dst, n);
  };
  cast(in_w,   in_w16,  2 * QKVDIM * DMODEL);
  cast(out_w,  out_w16, 2 * DMODEL * DMODEL);
  cast(lin1_w, l1_w16,  2 * FFDIM * DMODEL);
  cast(lin2_w, l2_w16,  2 * DMODEL * FFDIM);

  // ---- embedding ----
  embed_kernel<<<(MROWS * DMODEL + 255) / 256, 256, 0, stream>>>(
      row_ids, col_ids, values, row_tab, col_tab, val_w, val_b, xf, xh);

  const size_t attn_lds =
      (size_t)(16 * SEQ * 2 + 256 + 256 + 16 * 32) * sizeof(float);  // ~260KB

  for (int l = 0; l < 2; ++l) {
    // QKV projection: [16384,128] x [384,128]^T -> f16 qkv
    gemm_wmma_kernel<<<dim3(MROWS / 16, QKVDIM / 128), 256, 0, stream>>>(
        xh, in_w16 + (size_t)l * QKVDIM * DMODEL, in_b + l * QKVDIM,
        nullptr, qkvh, MROWS, QKVDIM, DMODEL, 0);

    // transpose V for contiguous WMMA B-fragments
    vtrans_kernel<<<(BATCH * NHEAD * DHEAD * SEQ + 255) / 256, 256, 0, stream>>>(
        qkvh, vT);

    // fused attention -> ctx16 (f16) + head-averaged attention output
    attn_fused_kernel<<<BATCH * (SEQ / 16), 256, attn_lds, stream>>>(
        qkvh, vT, ctx16, l == 0 ? attn0 : attn1);

    // output projection -> f32 ybuf
    gemm_wmma_kernel<<<dim3(MROWS / 16, 1), 256, 0, stream>>>(
        ctx16, out_w16 + (size_t)l * DMODEL * DMODEL, out_b + l * DMODEL,
        ybuf, nullptr, MROWS, DMODEL, DMODEL, 0);

    // x = LN(x + attn_out)
    add_ln_kernel<<<MROWS / 8, 256, 0, stream>>>(
        xf, ybuf, ln1_g + l * DMODEL, ln1_b + l * DMODEL, xf, xh);

    // FF1 + ReLU -> f16 ffh
    gemm_wmma_kernel<<<dim3(MROWS / 16, FFDIM / 128), 256, 0, stream>>>(
        xh, l1_w16 + (size_t)l * FFDIM * DMODEL, lin1_b + l * FFDIM,
        nullptr, ffh, MROWS, FFDIM, DMODEL, 1);

    // FF2 -> f32 ybuf
    gemm_wmma_kernel<<<dim3(MROWS / 16, 1), 256, 0, stream>>>(
        ffh, l2_w16 + (size_t)l * DMODEL * FFDIM, lin2_b + l * DMODEL,
        ybuf, nullptr, MROWS, DMODEL, FFDIM, 0);

    // x = LN(x + ff)
    add_ln_kernel<<<MROWS / 8, 256, 0, stream>>>(
        xf, ybuf, ln2_g + l * DMODEL, ln2_b + l * DMODEL, xf, xh);
  }

  // predictions
  predict_kernel<<<MROWS / 8, 256, 0, stream>>>(xf, outp_w, outp_b, pred);
}
